// StyleGAN2GeneratorBackbone_5480378269901
// MI455X (gfx1250) — matrix operand order repaired and checked by reading, not compiled, not run on hardware
//
#include <hip/hip_runtime.h>

typedef __bf16 bf16;
typedef __bf16 v16bf __attribute__((ext_vector_type(16)));
typedef __bf16 v8bf  __attribute__((ext_vector_type(8)));
typedef float  v8f   __attribute__((ext_vector_type(8)));
typedef unsigned int u32x4 __attribute__((ext_vector_type(4)));
typedef int          i32x4 __attribute__((ext_vector_type(4)));
typedef int          i32x8 __attribute__((ext_vector_type(8)));

#define BATCH 16
#define CONDD 512
#define EPSV  1e-8f

#if defined(__has_builtin)
#  if __has_builtin(__builtin_amdgcn_tensor_load_to_lds) && __has_builtin(__builtin_amdgcn_s_wait_tensorcnt)
#    define HAVE_TDM 1
#  endif
#endif
#ifndef HAVE_TDM
#  define HAVE_TDM 0
#endif

// ---------------------------------------------------------------------------
// style GEMM: s[b,n] = sum_k cond[b,k] * style[n,k]   (M=16, N=cin, K=512)
// one wave per 16x16 tile; bf16 WMMA, f32 accumulate; writes f32 (for demod)
// and bf16 (for conv modulation) copies of s.
// ---------------------------------------------------------------------------
__global__ __launch_bounds__(32)
void style_gemm_kernel(const float* __restrict__ cond,
                       const float* __restrict__ style,
                       float* __restrict__ s, bf16* __restrict__ sbf, int cin) {
  const int lane = threadIdx.x & 31;
  const int n0   = blockIdx.x * 16;
  const int row  = lane & 15;
  const int hi   = lane >> 4;
  v8f acc = {};
  for (int k0 = 0; k0 < CONDD; k0 += 32) {
    v16bf a, b;
    const float* ap = cond  + row * CONDD + k0 + hi * 16;
    const float* bp = style + (n0 + row) * CONDD + k0 + hi * 16;
#pragma unroll
    for (int t = 0; t < 8; ++t) {
      a[t]     = (bf16)ap[t];
      a[t + 8] = (bf16)ap[8 + t];
    }
#pragma unroll
    for (int t = 0; t < 16; ++t) b[t] = (bf16)bp[t];
    acc = __builtin_amdgcn_wmma_f32_16x16x32_bf16(false, a, false, b,
                                                  (short)0, acc, false, false);
  }
#pragma unroll
  for (int r = 0; r < 8; ++r) {
    int m = r + hi * 8;
    s[m * cin + n0 + row]   = acc[r];
    sbf[m * cin + n0 + row] = (bf16)acc[r];
  }
}

// Wsq[o,i] = sum_r W[o,i,r]^2
__global__ void wsq_kernel(const float* __restrict__ w, float* __restrict__ wsq, int n) {
  int t = blockIdx.x * blockDim.x + threadIdx.x;
  if (t >= n) return;
  const float* p = w + t * 9;
  float acc = 0.f;
#pragma unroll
  for (int r = 0; r < 9; ++r) acc += p[r] * p[r];
  wsq[t] = acc;
}

// d[b,o] = rsqrt(sum_i Wsq[o,i]*s[b,i]^2 + eps)  (exact f32)
__global__ void demod_kernel(const float* __restrict__ wsq, const float* __restrict__ s,
                             float* __restrict__ d, int cin, int cout) {
  int t = blockIdx.x * blockDim.x + threadIdx.x;
  if (t >= BATCH * cout) return;
  int b = t / cout, o = t - b * cout;
  const float* wr = wsq + o * cin;
  const float* sr = s + b * cin;
  float acc = 0.f;
  for (int i = 0; i < cin; ++i) { float sv = sr[i]; acc += wr[i] * sv * sv; }
  d[t] = rsqrtf(acc + EPSV);
}

// weight prep: wbf[o, r*cin+i] = (bf16) w[o, i*9+r]   (k-order = r*cin+i, bf16)
__global__ void wprep_kernel(const float* __restrict__ w, bf16* __restrict__ wbf,
                             int cin, int n) {
  int t = blockIdx.x * blockDim.x + threadIdx.x;
  if (t >= n) return;
  int Ktot = cin * 9;
  int o = t / Ktot; int k = t - o * Ktot;
  int r = k / cin;  int i = k - r * cin;
  wbf[t] = (bf16)w[o * Ktot + i * 9 + r];
}

// ---------------------------------------------------------------------------
// Modulated 3x3 conv, implicit GEMM, NHWC bf16 activations.
//   M = B*H*W, N = cout, K enumerated as r*cin+i (r = filter tap).
// 64x64 output tile per 256-thread block: 8 waves = 4 M-subtiles x 2 N-groups,
// each wave keeps 2 accumulators and issues 2 WMMAs per 32-wide K-chunk,
// reusing its A fragment (halves global x traffic vs a 32-wide N tile).
// A stage: contiguous 16B load + v_pk_mul_bf16 + 16B ds_store per thread.
// B stage: TDM tensor_load_to_lds (wave 0) of a 64x32 bf16 tile, or
//          contiguous per-thread 16B copies as fallback.
// ---------------------------------------------------------------------------
__global__ __launch_bounds__(256)
void modconv_kernel(const bf16* __restrict__ x, const bf16* __restrict__ wbf,
                    const bf16* __restrict__ sbf, const float* __restrict__ dmod,
                    const float* __restrict__ bias, const float* __restrict__ nscale,
                    const float* __restrict__ noise, bf16* __restrict__ y,
                    int cin, int cout, int H, int W) {
  __shared__ bf16 As[64 * 32];
  __shared__ bf16 Bs[64 * 32];
  const int tid  = threadIdx.x;
  const int lane = tid & 31;
  const int wave = tid >> 5;
  const int wm   = wave & 3;
  const int wn   = wave >> 2;
  const int m0   = blockIdx.x * 64;
  const int n0   = blockIdx.y * 64;
  const int HW   = H * W;
  const int Ktot = cin * 9;
  const int csteps = cin >> 5;

  // A staging role: one LDS row mm, 8 contiguous k within the chunk
  const int mm  = tid >> 2;
  const int kk0 = (tid & 3) << 3;
  const int m   = m0 + mm;
  const int b   = m / HW;
  const int pq  = m - b * HW;
  const int p   = pq / W, q = pq - p * W;
  const bf16* sprow = sbf + b * cin + kk0;
  bf16* adst = As + mm * 32 + kk0;

  // B staging role (fallback): one LDS row nn (of 64), 8 contiguous k
  bf16* bdst = Bs + mm * 32 + kk0;       // same (row, k-chunk) decomposition
  const bf16* bsrc_row = wbf + (size_t)(n0 + mm) * Ktot + kk0;

#if HAVE_TDM
  const unsigned ldsB = (unsigned)(uintptr_t)Bs;      // LDS byte offset (low 32 bits of flat)
#endif

  // fragment addresses (fixed)
  const bf16* arow  = As + (wm * 16 + (lane & 15)) * 32;
  const bf16* brow0 = Bs + (wn * 32 + (lane & 15)) * 32;
  const bf16* brow1 = brow0 + 16 * 32;
  const int abase = (lane >> 4) << 4;   // 0 or 16
  const int bbase = (lane >> 4) << 4;   // 0 or 16

  v8f acc0 = {};
  v8f acc1 = {};
  for (int r = 0; r < 9; ++r) {
    const int dy = r / 3, dx = r - dy * 3;
    const int py = p + dy - 1, px = q + dx - 1;
    const bool valid = (py >= 0) && (py < H) && (px >= 0) && (px < W);
    const bf16* xp = x + ((size_t)((b * H + py) * W + px)) * cin + kk0;
    const bf16* sp = sprow;
    for (int c = 0; c < csteps; ++c) {
      // ---- stage A (modulate + pad)
      v8bf xv = {};
      if (valid) xv = *(const v8bf*)xp;
      v8bf sv = *(const v8bf*)sp;
      *(v8bf*)adst = xv * sv;                          // v_pk_mul_bf16 x4
      // ---- stage B (64 rows x 32 k)
#if HAVE_TDM
      if (wave == 0) {
        unsigned long long ga =
            (unsigned long long)(uintptr_t)(wbf + (size_t)n0 * Ktot + r * cin + c * 32);
        u32x4 g0;
        g0[0] = 1u;                                            // count=1 (valid user D#)
        g0[1] = ldsB;                                          // lds_addr
        g0[2] = (unsigned)ga;                                  // global_addr[31:0]
        g0[3] = (unsigned)((ga >> 32) & 0x1FFFFFFull) | (2u << 30);  // addr[56:32] | type=2
        i32x8 g1;
        g1[0] = (1 << 16);                                     // data_size = 2B
        g1[1] = (int)(((unsigned)Ktot & 0xFFFFu) << 16);       // tensor_dim0[15:0]
        g1[2] = (int)((((unsigned)Ktot >> 16) & 0xFFFFu) | (64u << 16)); // dim0 hi | dim1=64
        g1[3] = (int)(32u << 16);                              // tile_dim0 = 32
        g1[4] = 64;                                            // tile_dim1 = 64, tile_dim2 = 0
        g1[5] = Ktot;                                          // dim0_stride[31:0]
        g1[6] = 0;
        g1[7] = 0;
        i32x4 g2 = {0, 0, 0, 0};
        i32x4 g3 = {0, 0, 0, 0};
#if __clang_major__ >= 23
        i32x8 g4 = {0, 0, 0, 0, 0, 0, 0, 0};
        __builtin_amdgcn_tensor_load_to_lds(g0, g1, g2, g3, g4, 0);
#else
        __builtin_amdgcn_tensor_load_to_lds(g0, g1, g2, g3, 0);
#endif
        __builtin_amdgcn_s_wait_tensorcnt(0);
      }
#else
      {
        const bf16* wp = bsrc_row + r * cin + c * 32;
        *(v8bf*)bdst = *(const v8bf*)wp;
        if (c + 1 < csteps) __builtin_prefetch(wp + 32, 0, 1);  // global_prefetch_b8
      }
#endif
      __syncthreads();
      // ---- fragments (two 16B ds loads each) + 2 WMMAs sharing the A fragment
      v8bf alo = *(const v8bf*)(arow + abase);
      v8bf ahi = *(const v8bf*)(arow + abase + 8);
      v16bf a = __builtin_shufflevector(alo, ahi, 0, 1, 2, 3, 4, 5, 6, 7,
                                        8, 9, 10, 11, 12, 13, 14, 15);
      v8bf b0lo = *(const v8bf*)(brow0 + bbase);
      v8bf b0hi = *(const v8bf*)(brow0 + bbase + 8);
      v16bf bb0 = __builtin_shufflevector(b0lo, b0hi, 0, 1, 2, 3, 4, 5, 6, 7,
                                          8, 9, 10, 11, 12, 13, 14, 15);
      acc0 = __builtin_amdgcn_wmma_f32_16x16x32_bf16(false, a, false, bb0,
                                                     (short)0, acc0, false, false);
      v8bf b1lo = *(const v8bf*)(brow1 + bbase);
      v8bf b1hi = *(const v8bf*)(brow1 + bbase + 8);
      v16bf bb1 = __builtin_shufflevector(b1lo, b1hi, 0, 1, 2, 3, 4, 5, 6, 7,
                                          8, 9, 10, 11, 12, 13, 14, 15);
      acc1 = __builtin_amdgcn_wmma_f32_16x16x32_bf16(false, a, false, bb1,
                                                     (short)0, acc1, false, false);
      __syncthreads();
      xp += 32;
      sp += 32;
    }
  }

  // ---- fused epilogue: demod, bias, noise; y is NHWC bf16
  const int nloc = lane & 15;
#pragma unroll
  for (int sub = 0; sub < 2; ++sub) {
    const int n = n0 + wn * 32 + sub * 16 + nloc;
    const float bi = bias[n];
    const float ns = nscale[n];
    const v8f accv = sub ? acc1 : acc0;
#pragma unroll
    for (int r = 0; r < 8; ++r) {
      int mloc = r + ((lane >> 4) << 3);
      int me = m0 + wm * 16 + mloc;
      int be = me / HW; int pqe = me - be * HW;
      float dv = dmod[be * cout + n];
      size_t ni = (size_t)(be * cout + n) * HW + pqe;        // noise is NCHW f32
      y[(size_t)me * cout + n] = (bf16)(accv[r] * dv + bi + noise[ni] * ns);
    }
  }
}

// to-RGB 1x1 conv accumulate (x: NHWC bf16; rgb: NCHW f32)
__global__ void rgb_kernel(const bf16* __restrict__ x, const float* __restrict__ rw,
                           const float* __restrict__ rb, const float* __restrict__ src,
                           float* __restrict__ dst, int cout, int HW) {
  int t = blockIdx.x * blockDim.x + threadIdx.x;
  if (t >= BATCH * HW) return;
  int b = t / HW, pq = t - b * HW;
  const bf16* xp = x + ((size_t)b * HW + pq) * cout;
  float a0 = 0.f, a1 = 0.f, a2 = 0.f;
  for (int i = 0; i < cout; ++i) {
    float v = (float)xp[i];
    a0 += v * rw[i];
    a1 += v * rw[cout + i];
    a2 += v * rw[2 * cout + i];
  }
  int base = b * 3 * HW + pq;
  dst[base]          = src[base]          + a0 + rb[0];
  dst[base + HW]     = src[base + HW]     + a1 + rb[1];
  dst[base + 2 * HW] = src[base + 2 * HW] + a2 + rb[2];
}

// nearest 2x upsample, NHWC bf16 activations
__global__ void up2_nhwc_kernel(const bf16* __restrict__ in, bf16* __restrict__ out,
                                int C, int H, int W) {
  int t = blockIdx.x * blockDim.x + threadIdx.x;
  int W2 = W * 2, H2 = H * 2;
  int n = BATCH * C * H2 * W2;
  if (t >= n) return;
  int c = t % C; int tmp = t / C;
  int w2 = tmp % W2; tmp /= W2;
  int h2 = tmp % H2; int b = tmp / H2;
  out[t] = in[((size_t)((b * H + (h2 >> 1)) * W) + (w2 >> 1)) * C + c];
}

// nearest 2x upsample, NCHW f32 (rgb)
__global__ void up2_nchw_kernel(const float* __restrict__ in, float* __restrict__ out,
                                int C, int H, int W) {
  int t = blockIdx.x * blockDim.x + threadIdx.x;
  int H2 = H * 2, W2 = W * 2;
  int n = BATCH * C * H2 * W2;
  if (t >= n) return;
  int w2 = t % W2; int tmp = t / W2;
  int h2 = tmp % H2; int bc = tmp / H2;
  out[t] = in[(bc * H + (h2 >> 1)) * W + (w2 >> 1)];
}

// broadcast base_map [512,4,4] NCHW f32 -> [B,4,4,512] NHWC bf16
__global__ void bcast_nhwc_kernel(const float* __restrict__ base, bf16* __restrict__ out) {
  int t = blockIdx.x * blockDim.x + threadIdx.x;
  if (t >= BATCH * 16 * 512) return;
  int c = t % 512; int tmp = t / 512; int pq = tmp % 16;
  out[t] = (bf16)base[c * 16 + pq];
}

__global__ void zero_kernel(float* __restrict__ p, int n) {
  int t = blockIdx.x * blockDim.x + threadIdx.x;
  if (t < n) p[t] = 0.f;
}

// ---------------------------------------------------------------------------
// Input flattening (dict insertion order):
//  [0] condition  [1] base_map
//  [2+blk*10 +0..9]: style0,style1,W0,W1,b0,b1,nscale0,nscale1,rgb_W,rgb_b
//  [62+blk*2+j]: noise[blk][j]
// ---------------------------------------------------------------------------
extern "C" void kernel_launch(void* const* d_in, const int* in_sizes, int n_in,
                              void* d_out, int out_size, void* d_ws, size_t ws_size,
                              hipStream_t stream) {
  (void)in_sizes; (void)n_in; (void)out_size; (void)ws_size;
  static const int factors[7] = {4, 4, 4, 2, 2, 1, 1};
  const float* condition = (const float*)d_in[0];
  const float* base_map  = (const float*)d_in[1];

  char* pws = (char*)d_ws;
  const size_t ACT_BYTES = (size_t)BATCH * 128 * 128 * 128 * 2;  // 64 MiB bf16
  const size_t RGB_BYTES = (size_t)BATCH * 3 * 128 * 128 * 4;
  bf16*  actA = (bf16*)pws;  pws += ACT_BYTES;
  bf16*  actB = (bf16*)pws;  pws += ACT_BYTES;
  float* rgbA = (float*)pws; pws += RGB_BYTES;
  float* rgbB = (float*)pws; pws += RGB_BYTES;
  float* sbuf = (float*)pws; pws += (size_t)BATCH * 512 * 4;
  bf16*  sbf  = (bf16*)pws;  pws += (size_t)BATCH * 512 * 2;
  float* dbuf = (float*)pws; pws += (size_t)BATCH * 512 * 4;
  float* wsq  = (float*)pws; pws += (size_t)512 * 512 * 4;
  bf16*  wbf  = (bf16*)pws;  pws += (size_t)512 * 4608 * 2;

  bf16* cur = actA; bf16* alt = actB;
  float* rgb = rgbA; float* rgbAlt = rgbB;

  {
    int total = BATCH * 16 * 512;
    bcast_nhwc_kernel<<<(total + 255) / 256, 256, 0, stream>>>(base_map, cur);
    int nr = BATCH * 3 * 16;
    zero_kernel<<<(nr + 255) / 256, 256, 0, stream>>>(rgb, nr);
  }

  int res = 4;
  for (int blk = 0; blk < 6; ++blk) {
    const int cin  = factors[blk] * 128;
    const int cout = factors[blk + 1] * 128;
    const int HW   = res * res;
    const int base = 2 + blk * 10;

    for (int j = 0; j < 2; ++j) {
      const int co = (j == 0) ? cin : cout;
      const float* style = (const float*)d_in[base + j];
      const float* W     = (const float*)d_in[base + 2 + j];
      const float* bias  = (const float*)d_in[base + 4 + j];
      const float* nsc   = (const float*)d_in[base + 6 + j];
      const float* noise = (const float*)d_in[62 + blk * 2 + j];

      style_gemm_kernel<<<cin / 16, 32, 0, stream>>>(condition, style, sbuf, sbf, cin);
      int nw = co * cin;
      wsq_kernel<<<(nw + 255) / 256, 256, 0, stream>>>(W, wsq, nw);
      int nd = BATCH * co;
      demod_kernel<<<(nd + 255) / 256, 256, 0, stream>>>(wsq, sbuf, dbuf, cin, co);
      int nt = co * cin * 9;
      wprep_kernel<<<(nt + 255) / 256, 256, 0, stream>>>(W, wbf, cin, nt);

      dim3 grid(BATCH * HW / 64, co / 64);
      modconv_kernel<<<grid, 256, 0, stream>>>(cur, wbf, sbf, dbuf, bias, nsc,
                                               noise, alt, cin, co, res, res);
      { bf16* t = cur; cur = alt; alt = t; }
    }

    const float* rgbW = (const float*)d_in[base + 8];
    const float* rgbb = (const float*)d_in[base + 9];
    float* dst = (blk == 5) ? (float*)d_out : rgb;
    int n = BATCH * HW;
    rgb_kernel<<<(n + 255) / 256, 256, 0, stream>>>(cur, rgbW, rgbb, rgb, dst, cout, HW);

    if (blk < 5) {
      int n2 = BATCH * cout * 4 * HW;
      up2_nhwc_kernel<<<(n2 + 255) / 256, 256, 0, stream>>>(cur, alt, cout, res, res);
      { bf16* t = cur; cur = alt; alt = t; }
      int n3 = BATCH * 3 * 4 * HW;
      up2_nchw_kernel<<<(n3 + 255) / 256, 256, 0, stream>>>(rgb, rgbAlt, 3, res, res);
      { float* t = rgb; rgb = rgbAlt; rgbAlt = t; }
      res *= 2;
    }
  }
}
